// VisionTokenMerger_29575144800699
// MI455X (gfx1250) — matrix operand order repaired and checked by reading, not compiled, not run on hardware
//
#include <hip/hip_runtime.h>
#include <hip/hip_bf16.h>

typedef __attribute__((ext_vector_type(16))) _Float16 v16h;
typedef __attribute__((ext_vector_type(8)))  _Float16 v8h;
typedef __attribute__((ext_vector_type(8)))  float    v8f;

#define NB   64     // batches
#define NT   1024   // tokens
#define MC   64     // metric channels
#define XD   768    // x channels
#define HT   512    // half tokens (even/odd split)
#define RR   256    // merge count

// Order-preserving float <-> uint32 maps (for branchless max+argmax keys).
__device__ __forceinline__ unsigned f2ord(float f) {
  unsigned u = __float_as_uint(f);
  return (u & 0x80000000u) ? ~u : (u | 0x80000000u);
}
__device__ __forceinline__ float ord2f(unsigned o) {
  unsigned u = (o & 0x80000000u) ? (o ^ 0x80000000u) : ~o;
  return __uint_as_float(u);
}

// ---------------------------------------------------------------------------
// Kernel 0a: normalize metric rows and store as f16 (one pass, 16MB->8MB).
// ---------------------------------------------------------------------------
__global__ __launch_bounds__(256) void k_norm16(const float* __restrict__ metric,
                                                _Float16* __restrict__ m16) {
  const size_t tok = (size_t)blockIdx.x * 256 + threadIdx.x;   // 0 .. NB*NT-1
  const float4* row = (const float4*)(metric + tok * MC);
  float4 v[MC / 4];
  float s = 0.f;
  for (int q = 0; q < MC / 4; ++q) {
    v[q] = row[q];
    s += v[q].x * v[q].x + v[q].y * v[q].y + v[q].z * v[q].z + v[q].w * v[q].w;
  }
  const float inv = rsqrtf(fmaxf(s, 1e-30f));
  _Float16* out = m16 + tok * MC;
  for (int q = 0; q < MC / 4; ++q) {
    out[4 * q + 0] = (_Float16)(v[q].x * inv);
    out[4 * q + 1] = (_Float16)(v[q].y * inv);
    out[4 * q + 2] = (_Float16)(v[q].z * inv);
    out[4 * q + 3] = (_Float16)(v[q].w * inv);
  }
}

// ---------------------------------------------------------------------------
// Kernel 0b (fallback path): per-token inverse L2 norm only.
// ---------------------------------------------------------------------------
__global__ __launch_bounds__(256) void k_norms(const float* __restrict__ metric,
                                               float* __restrict__ invNorm) {
  const size_t tok = (size_t)blockIdx.x * 256 + threadIdx.x;
  const float4* row = (const float4*)(metric + tok * MC);
  float s = 0.f;
  for (int q = 0; q < MC / 4; ++q) {
    float4 v = row[q];
    s += v.x * v.x + v.y * v.y + v.z * v.z + v.w * v.w;
  }
  invNorm[tok] = rsqrtf(fmaxf(s, 1e-30f));
}

// ---------------------------------------------------------------------------
// Shared epilogue: per-lane running keys -> butterfly -> write node max/idx.
// ---------------------------------------------------------------------------
__device__ __forceinline__ void reduce_and_write(unsigned long long* runKey,
                                                 int b, int mt, int M, int hi,
                                                 float* nodeMax, int* nodeIdx) {
  for (int r = 0; r < 8; ++r) {
    unsigned long long k = runKey[r];
    for (int m = 1; m < 16; m <<= 1) {
      unsigned olo = __shfl_xor((unsigned)k, m, 32);
      unsigned ohi = __shfl_xor((unsigned)(k >> 32), m, 32);
      unsigned long long ok = ((unsigned long long)ohi << 32) | olo;
      k = (ok > k) ? ok : k;
    }
    runKey[r] = k;
  }
  if (M == 0) {  // lanes 0 (rows 0..7) and 16 (rows 8..15) write
    int rowBase = b * HT + 16 * mt + 8 * hi;
    for (int r = 0; r < 8; ++r) {
      nodeMax[rowBase + r] = ord2f((unsigned)(runKey[r] >> 32));
      nodeIdx[rowBase + r] = (int)~(unsigned)runKey[r];
    }
  }
}

// ---------------------------------------------------------------------------
// Kernel 1 (fast path): WMMA GEMM on pre-normalized f16 metric.
// Fragments load as two aligned b128s each -- zero per-element VALU.
// ---------------------------------------------------------------------------
__global__ __launch_bounds__(32) void k_scores_f16(const _Float16* __restrict__ m16,
                                                   float* __restrict__ nodeMax,
                                                   int*   __restrict__ nodeIdx) {
  const int lane = threadIdx.x;        // 0..31
  const int b    = blockIdx.y;
  const int mt   = blockIdx.x;         // row tile 0..31
  const int M    = lane & 15;
  const int hi   = lane >> 4;
  const _Float16* mbase = m16 + (size_t)b * NT * MC;

  // A fragments: a0 = row[hi*8 .. +7] ++ row[hi*8+16 .. +7]; a1 = same +32.
  v16h a0, a1;
  {
    const _Float16* arow = mbase + (size_t)(2 * (16 * mt + M)) * MC;
    v8h lo0 = *(const v8h*)(arow + (hi << 3));
    v8h hi0 = *(const v8h*)(arow + (hi << 3) + 16);
    v8h lo1 = *(const v8h*)(arow + (hi << 3) + 32);
    v8h hi1 = *(const v8h*)(arow + (hi << 3) + 48);
    a0 = __builtin_shufflevector(lo0, hi0, 0,1,2,3,4,5,6,7,8,9,10,11,12,13,14,15);
    a1 = __builtin_shufflevector(lo1, hi1, 0,1,2,3,4,5,6,7,8,9,10,11,12,13,14,15);
  }

  unsigned long long runKey[8];
  for (int r = 0; r < 8; ++r) runKey[r] = 0ull;

  for (int nt = 0; nt < 32; ++nt) {
    // B fragments: b0 = col-row[hi*8 .. +7] ++ col-row[hi*8+16 .. +7]; b1 = same +32.
    const _Float16* brow = mbase + (size_t)(2 * (16 * nt + M) + 1) * MC;
    v8h lo0 = *(const v8h*)(brow + (hi << 3));
    v8h hi0 = *(const v8h*)(brow + (hi << 3) + 16);
    v8h lo1 = *(const v8h*)(brow + (hi << 3) + 32);
    v8h hi1 = *(const v8h*)(brow + (hi << 3) + 48);
    v16h b0 = __builtin_shufflevector(lo0, hi0, 0,1,2,3,4,5,6,7,8,9,10,11,12,13,14,15);
    v16h b1 = __builtin_shufflevector(lo1, hi1, 0,1,2,3,4,5,6,7,8,9,10,11,12,13,14,15);

    v8f c = {};
    c = __builtin_amdgcn_wmma_f32_16x16x32_f16(false, a0, false, b0, (short)0, c, false, false);
    c = __builtin_amdgcn_wmma_f32_16x16x32_f16(false, a1, false, b1, (short)0, c, false, false);

    const unsigned invIdx = ~(unsigned)(16 * nt + M);
    for (int r = 0; r < 8; ++r) {
      unsigned long long key = ((unsigned long long)f2ord(c[r]) << 32) | invIdx;
      runKey[r] = (key > runKey[r]) ? key : runKey[r];
    }
  }
  reduce_and_write(runKey, b, mt, M, hi, nodeMax, nodeIdx);
}

// ---------------------------------------------------------------------------
// Kernel 1 (fallback path): WMMA GEMM with on-the-fly normalize/convert.
// ---------------------------------------------------------------------------
__global__ __launch_bounds__(32) void k_scores_f32(const float* __restrict__ metric,
                                                   const float* __restrict__ invNorm,
                                                   float* __restrict__ nodeMax,
                                                   int*   __restrict__ nodeIdx) {
  const int lane = threadIdx.x;
  const int b    = blockIdx.y;
  const int mt   = blockIdx.x;
  const int M    = lane & 15;
  const int hi   = lane >> 4;
  const float* mbase = metric  + (size_t)b * NT * MC;
  const float* nbase = invNorm + (size_t)b * NT;

  v16h a0, a1;
  {
    const int tokA = 2 * (16 * mt + M);
    const float* arow = mbase + (size_t)tokA * MC;
    const float ia = nbase[tokA];
    for (int j = 0; j < 16; ++j) {
      int k = (hi << 3) + (j & 7) + ((j >> 3) << 4);
      a0[j] = (_Float16)(arow[k]      * ia);
      a1[j] = (_Float16)(arow[32 + k] * ia);
    }
  }

  unsigned long long runKey[8];
  for (int r = 0; r < 8; ++r) runKey[r] = 0ull;

  for (int nt = 0; nt < 32; ++nt) {
    const int tokB = 2 * (16 * nt + M) + 1;
    const float* brow = mbase + (size_t)tokB * MC;
    const float ib = nbase[tokB];
    v16h b0, b1;
    for (int j = 0; j < 16; ++j) {
      int k = (hi << 3) + (j & 7) + ((j >> 3) << 4);
      b0[j] = (_Float16)(brow[k]      * ib);
      b1[j] = (_Float16)(brow[32 + k] * ib);
    }
    v8f c = {};
    c = __builtin_amdgcn_wmma_f32_16x16x32_f16(false, a0, false, b0, (short)0, c, false, false);
    c = __builtin_amdgcn_wmma_f32_16x16x32_f16(false, a1, false, b1, (short)0, c, false, false);

    const unsigned invIdx = ~(unsigned)(16 * nt + M);
    for (int r = 0; r < 8; ++r) {
      unsigned long long key = ((unsigned long long)f2ord(c[r]) << 32) | invIdx;
      runKey[r] = (key > runKey[r]) ? key : runKey[r];
    }
  }
  reduce_and_write(runKey, b, mt, M, hi, nodeMax, nodeIdx);
}

// ---------------------------------------------------------------------------
// Kernel 2: stable descending rank sort (== argsort(-node_max)) per batch.
// ---------------------------------------------------------------------------
__global__ __launch_bounds__(512) void k_rank(const float* __restrict__ nodeMax,
                                              int* __restrict__ edgeIdx) {
  __shared__ float vals[HT];
  const int b = blockIdx.x, t = threadIdx.x;
  vals[t] = nodeMax[b * HT + t];
  __syncthreads();
  const float v = vals[t];
  int rank = 0;
  for (int j = 0; j < HT; ++j) {
    float u = vals[j];
    rank += (u > v) || (u == v && j < t);
  }
  edgeIdx[b * HT + rank] = t;
}

// ---------------------------------------------------------------------------
// Kernel 3: counts = 1 + histogram(dst_idx); also writes merged_sizes.
// ---------------------------------------------------------------------------
__global__ __launch_bounds__(512) void k_counts(const int* __restrict__ edgeIdx,
                                                const int* __restrict__ nodeIdx,
                                                float* __restrict__ counts,
                                                float* __restrict__ outSizes) {
  __shared__ float cnt[HT];
  const int b = blockIdx.x, t = threadIdx.x;
  cnt[t] = 1.0f;
  __syncthreads();
  if (t < RR) {
    int s = edgeIdx[b * HT + t];       // src_idx
    int d = nodeIdx[b * HT + s];       // dst_idx
    atomicAdd(&cnt[d], 1.0f);
  }
  __syncthreads();
  counts[b * HT + t] = cnt[t];
  outSizes[(size_t)b * XD + RR + t] = cnt[t];
  if (t < RR) outSizes[(size_t)b * XD + t] = 1.0f;
}

// ---------------------------------------------------------------------------
// Kernel 4a: base rows of merged_tokens (float4 wide: b128 traffic).
// ---------------------------------------------------------------------------
__global__ __launch_bounds__(192) void k_base(const float* __restrict__ x,
                                              const int* __restrict__ edgeIdx,
                                              float* __restrict__ outTok) {
  const int b = blockIdx.y;
  const int row = blockIdx.x;           // 0..767
  int tok;
  if (row < RR) tok = 2 * edgeIdx[b * HT + RR + row];   // unm: even tokens
  else          tok = 2 * (row - RR) + 1;               // dst: odd tokens
  const float4* srcRow = (const float4*)(x + ((size_t)b * NT + tok) * XD);
  float4* dstRow = (float4*)(outTok + ((size_t)b * XD + row) * XD);
  dstRow[threadIdx.x] = srcRow[threadIdx.x];            // 192 float4 = 768 f32
}

// ---------------------------------------------------------------------------
// Kernel 4b: scatter-add selected src rows onto their dst rows.
// ---------------------------------------------------------------------------
__global__ __launch_bounds__(256) void k_scatter(const float* __restrict__ x,
                                                 const int* __restrict__ edgeIdx,
                                                 const int* __restrict__ nodeIdx,
                                                 float* __restrict__ outTok) {
  const int b = blockIdx.y;
  const int j = blockIdx.x;             // 0..255
  const int s = edgeIdx[b * HT + j];
  const int d = nodeIdx[b * HT + s];
  const float* srcRow = x + ((size_t)b * NT + 2 * s) * XD;
  float* dstRow = outTok + ((size_t)b * XD + RR + d) * XD;
  for (int c = threadIdx.x; c < XD; c += 256)
    atomicAdd(&dstRow[c], srcRow[c]);
}

// ---------------------------------------------------------------------------
// Kernel 4c: divide merged dst rows by counts (float4 wide).
// ---------------------------------------------------------------------------
__global__ __launch_bounds__(192) void k_div(float* __restrict__ outTok,
                                             const float* __restrict__ counts) {
  const int b = blockIdx.y;
  const int k = blockIdx.x;             // 0..511
  const float inv = 1.0f / counts[b * HT + k];
  float4* row = (float4*)(outTok + ((size_t)b * XD + RR + k) * XD);
  float4 v = row[threadIdx.x];
  v.x *= inv; v.y *= inv; v.z *= inv; v.w *= inv;
  row[threadIdx.x] = v;
}

extern "C" void kernel_launch(void* const* d_in, const int* in_sizes, int n_in,
                              void* d_out, int out_size, void* d_ws, size_t ws_size,
                              hipStream_t stream) {
  const float* metric = (const float*)d_in[0];   // (64,1024,64) f32
  const float* x      = (const float*)d_in[1];   // (64,1024,768) f32

  float* outTok   = (float*)d_out;                        // (64,768,768)
  float* outSizes = outTok + (size_t)NB * XD * XD;        // (64,768)

  // Workspace layout: 4 index/score arrays (512 KB) + metric staging.
  float* nodeMax = (float*)d_ws;                          // 64*512 f32
  int*   nodeIdx = (int*)(nodeMax + NB * HT);             // 64*512 i32
  int*   edgeIdx = nodeIdx + NB * HT;                     // 64*512 i32
  float* counts  = (float*)(edgeIdx + NB * HT);           // 64*512 f32
  char*  stage   = (char*)(counts + NB * HT);

  const size_t baseBytes = (size_t)4 * NB * HT * 4;
  const size_t f16Bytes  = (size_t)NB * NT * MC * sizeof(_Float16);   // 8 MB

  if (ws_size >= baseBytes + f16Bytes) {
    _Float16* m16 = (_Float16*)stage;
    k_norm16     <<<dim3(NB * NT / 256), 256, 0, stream>>>(metric, m16);
    k_scores_f16 <<<dim3(32, NB), 32, 0, stream>>>(m16, nodeMax, nodeIdx);
  } else {
    float* invNorm = (float*)stage;                       // 64*1024 f32
    k_norms      <<<dim3(NB * NT / 256), 256, 0, stream>>>(metric, invNorm);
    k_scores_f32 <<<dim3(32, NB), 32, 0, stream>>>(metric, invNorm, nodeMax, nodeIdx);
  }

  k_rank   <<<NB,           512, 0, stream>>>(nodeMax, edgeIdx);
  k_counts <<<NB,           512, 0, stream>>>(edgeIdx, nodeIdx, counts, outSizes);
  k_base   <<<dim3(XD, NB), 192, 0, stream>>>(x, edgeIdx, outTok);
  k_scatter<<<dim3(RR, NB), 256, 0, stream>>>(x, edgeIdx, nodeIdx, outTok);
  k_div    <<<dim3(HT, NB), 192, 0, stream>>>(outTok, counts);
}
